// TemporalAttention_16758962389416
// MI455X (gfx1250) — matrix operand order repaired and checked by reading, not compiled, not run on hardware
//
#include <hip/hip_runtime.h>

typedef __attribute__((ext_vector_type(16))) _Float16 v16h;
typedef __attribute__((ext_vector_type(8)))  float    v8f;
typedef __attribute__((ext_vector_type(4)))  _Float16 h4;

union alignas(32) F16x16 {
    v16h  v;
    _Float16 h[16];
    uint4 q[2];
};

__device__ __forceinline__ v8f zero8() {
    v8f z = {0.f, 0.f, 0.f, 0.f, 0.f, 0.f, 0.f, 0.f};
    return z;
}

// A-matrix fragment (16x32 f16): lane holds row M = lane%16.
// Chunks at k = kblock + 8*hi + {0..7} and kblock + 8*hi + 16 + {0..7}.
__device__ __forceinline__ v16h load_a_frag(const _Float16* __restrict__ row,
                                            int kblock, int hi) {
    F16x16 r;
    const _Float16* p = row + kblock + 8 * hi;
    r.q[0] = *(const uint4*)(p);
    r.q[1] = *(const uint4*)(p + 16);
    return r.v;
}

// B-matrix fragment (32x16 f16): lane holds column N = lane%16.
// Chunks at k = kblock + 8*hi + {0..7} and kblock + 8*hi + 16 + {0..7}.
__device__ __forceinline__ v16h load_b_frag(const _Float16* __restrict__ btrow,
                                            int kblock, int hi) {
    F16x16 r;
    const _Float16* p = btrow + kblock + 8 * hi;
    r.q[0] = *(const uint4*)(p);
    r.q[1] = *(const uint4*)(p + 16);
    return r.v;
}

__device__ __forceinline__ v8f wmma_f16(v16h a, v16h b, v8f c) {
    return __builtin_amdgcn_wmma_f32_16x16x32_f16(false, a, false, b,
                                                  (short)0, c, false, false);
}

// ---------------------------------------------------------------- converts
__global__ void cvt_f16x4_kernel(const float4* __restrict__ in,
                                 h4* __restrict__ out, int n4) {
    int i = blockIdx.x * blockDim.x + threadIdx.x;
    if (i < n4) {
        float4 v = in[i];
        h4 o = {(_Float16)v.x, (_Float16)v.y, (_Float16)v.z, (_Float16)v.w};
        out[i] = o;
    }
}

// in: K x N (f32), out: N x K (f16)  -- coalesced on output
__global__ void cvt_transpose_kernel(const float* __restrict__ in,
                                     _Float16* __restrict__ out, int K, int N) {
    int idx = blockIdx.x * blockDim.x + threadIdx.x;
    if (idx >= K * N) return;
    int k = idx % K;
    int n = idx / K;
    out[idx] = (_Float16)in[(size_t)k * N + n];
}

// ---------------------------------------------------------------- GEMM
// C[M,N] = A[M,K] (f16 row-major) * B (given as B^T: N x K f16 row-major) + bias
// Workgroup: 4 waves, 128(M) x 64(N) tile; wave w owns rows [32w, 32w+32):
// 2 A-fragments reuse each of the 4 B-fragments -> 8 WMMA per 12 loads.
template <bool OUT_HALF>
__global__ __launch_bounds__(128) void wmma_gemm_kernel(
        const _Float16* __restrict__ A, const _Float16* __restrict__ BT,
        const float* __restrict__ bias, void* __restrict__ Cout,
        int M, int N, int K) {
    const int lane  = threadIdx.x & 31;
    const int wave  = threadIdx.x >> 5;
    const int hi    = lane >> 4;
    const int ln    = lane & 15;
    const int mbase = blockIdx.x * 128 + wave * 32;
    const int nbase = blockIdx.y * 64;

    v8f acc[2][4];
#pragma unroll
    for (int mt = 0; mt < 2; ++mt)
#pragma unroll
        for (int nt = 0; nt < 4; ++nt) acc[mt][nt] = zero8();

    const _Float16* arow0 = A + (size_t)(mbase + ln) * K;
    const _Float16* arow1 = A + (size_t)(mbase + 16 + ln) * K;
    const _Float16* brow0 = BT + (size_t)(nbase + ln) * K;

    for (int k = 0; k < K; k += 32) {
        __builtin_prefetch(arow0 + k + 64, 0, 3);
        __builtin_prefetch(arow1 + k + 64, 0, 3);
        v16h a0 = load_a_frag(arow0, k, hi);
        v16h a1 = load_a_frag(arow1, k, hi);
#pragma unroll
        for (int nt = 0; nt < 4; ++nt) {
            v16h b = load_b_frag(brow0 + (size_t)(nt * 16) * K, k, hi);
            acc[0][nt] = wmma_f16(a0, b, acc[0][nt]);
            acc[1][nt] = wmma_f16(a1, b, acc[1][nt]);
        }
    }

#pragma unroll
    for (int nt = 0; nt < 4; ++nt) {
        int col = nbase + nt * 16 + ln;
        float bv = bias[col];
#pragma unroll
        for (int mt = 0; mt < 2; ++mt) {
#pragma unroll
            for (int r = 0; r < 8; ++r) {
                int row = mbase + mt * 16 + r + 8 * hi;
                float v = acc[mt][nt][r] + bv;
                if (OUT_HALF)
                    ((_Float16*)Cout)[(size_t)row * N + col] = (_Float16)v;
                else
                    ((float*)Cout)[(size_t)row * N + col] = v;
            }
        }
    }
}

// ---------------------------------------------------------------- attention
#define HWDIM 1024
#define CDIM  512
#define LDQKV 1536
#define HD    64

// grid (HW/64, T, H), block 128 (4 waves). Flash-style online softmax,
// k-block = 64. Row sums computed with a ones-matrix WMMA (rowsum(P) = P @ 1)
// instead of lane shuffles. AO written in the reference's transpose/reshape
// ("bug preserved") flat order: idx = h<<20 | q<<10 | t<<6 | d.
__global__ __launch_bounds__(128) void attn_kernel(
        const _Float16* __restrict__ qkv, _Float16* __restrict__ AO) {
    __shared__ _Float16 Vt[64][72];      // V^T tile [d][k], shared by the WG
    __shared__ _Float16 Pb[4][16][72];   // per-wave P scratch (16 x 64 + pad)

    const int tid   = threadIdx.x;
    const int lane  = tid & 31;
    const int wave  = tid >> 5;
    const int hi    = lane >> 4;
    const int ln    = lane & 15;
    const int h     = blockIdx.z;
    const int t     = blockIdx.y;
    const int qbase = blockIdx.x * 64;

    // Q fragments for this wave's 16 rows (loaded once, d = 0..63)
    const _Float16* qrow =
        qkv + (size_t)(t * HWDIM + qbase + wave * 16 + ln) * LDQKV + h * HD;
    v16h aq0 = load_a_frag(qrow, 0, hi);
    v16h aq1 = load_a_frag(qrow, 32, hi);

    // all-ones B fragment for row-sum WMMA
    F16x16 onesu;
#pragma unroll
    for (int i = 0; i < 16; ++i) onesu.h[i] = (_Float16)1.0f;
    const v16h bones = onesu.v;

    v8f o[4];
    float m[8], l[8], alpha[8];
#pragma unroll
    for (int nt = 0; nt < 4; ++nt) o[nt] = zero8();
#pragma unroll
    for (int r = 0; r < 8; ++r) { m[r] = -1e30f; l[r] = 0.f; }

    const float scale = 0.125f;  // 1/sqrt(64)

    for (int kb = 0; kb < HWDIM; kb += 64) {
        __syncthreads();  // all waves done reading previous Vt
        {   // stage V block transposed: 64 rows x 64 d -> Vt[d][klocal]
            int r    = tid >> 1;          // 0..63
            int dseg = (tid & 1) * 32;    // 0 or 32
            const _Float16* vp =
                qkv + (size_t)(t * HWDIM + kb + r) * LDQKV + 2 * CDIM + h * HD + dseg;
            F16x16 t0, t1;
            t0.q[0] = *(const uint4*)(vp);
            t0.q[1] = *(const uint4*)(vp + 8);
            t1.q[0] = *(const uint4*)(vp + 16);
            t1.q[1] = *(const uint4*)(vp + 24);
#pragma unroll
            for (int i = 0; i < 16; ++i) Vt[dseg + i][r] = t0.h[i];
#pragma unroll
            for (int i = 0; i < 16; ++i) Vt[dseg + 16 + i][r] = t1.h[i];
        }
        __syncthreads();

        // S = Q K^T for four 16-col sub-tiles (K rows are contiguous over d =
        // exactly the WMMA B-layout, straight from global / WGP$)
        v8f s[4];
#pragma unroll
        for (int sub = 0; sub < 4; ++sub) {
            int col = kb + sub * 16 + ln;
            const _Float16* krow =
                qkv + (size_t)(t * HWDIM + col) * LDQKV + CDIM + h * HD;
            v16h bk0 = load_b_frag(krow, 0, hi);
            v16h bk1 = load_b_frag(krow, 32, hi);
            v8f a = zero8();
            a = wmma_f16(aq0, bk0, a);
            a = wmma_f16(aq1, bk1, a);
            s[sub] = a;
        }

        // online max; row (r + 8*hi) lives across 16 lanes of one half
#pragma unroll
        for (int r = 0; r < 8; ++r) {
            float s0 = s[0][r] * scale;
            float s1 = s[1][r] * scale;
            float s2 = s[2][r] * scale;
            float s3 = s[3][r] * scale;
            float vmax = fmaxf(fmaxf(s0, s1), fmaxf(s2, s3));
#pragma unroll
            for (int msk = 8; msk >= 1; msk >>= 1)
                vmax = fmaxf(vmax, __shfl_xor(vmax, msk, 32));
            float mn = fmaxf(m[r], vmax);
            alpha[r] = __expf(m[r] - mn);
            m[r] = mn;
            int row = r + 8 * hi;
            Pb[wave][row][ln]      = (_Float16)__expf(s0 - mn);
            Pb[wave][row][16 + ln] = (_Float16)__expf(s1 - mn);
            Pb[wave][row][32 + ln] = (_Float16)__expf(s2 - mn);
            Pb[wave][row][48 + ln] = (_Float16)__expf(s3 - mn);
        }

        // reload P in A-fragment layout (same-wave LDS is in-order)
        v16h ap0 = load_a_frag(&Pb[wave][ln][0], 0, hi);
        v16h ap1 = load_a_frag(&Pb[wave][ln][0], 32, hi);

        // row sums of the f16-rounded P via ones-matrix WMMA
        v8f rs = zero8();
        rs = wmma_f16(ap0, bones, rs);
        rs = wmma_f16(ap1, bones, rs);

#pragma unroll
        for (int r = 0; r < 8; ++r) {
            l[r] = l[r] * alpha[r] + rs[r];
#pragma unroll
            for (int nt = 0; nt < 4; ++nt) o[nt][r] *= alpha[r];
        }

        // O += P @ V
#pragma unroll
        for (int nt = 0; nt < 4; ++nt) {
            v16h bv0 = load_b_frag(&Vt[nt * 16 + ln][0], 0, hi);
            v16h bv1 = load_b_frag(&Vt[nt * 16 + ln][0], 32, hi);
            o[nt] = wmma_f16(ap0, bv0, o[nt]);
            o[nt] = wmma_f16(ap1, bv1, o[nt]);
        }
    }

    // normalize and store with the reference's transpose/reshape mapping
#pragma unroll
    for (int r = 0; r < 8; ++r) {
        float inv = 1.0f / l[r];
        int q = qbase + wave * 16 + r + 8 * hi;
#pragma unroll
        for (int nt = 0; nt < 4; ++nt) {
            int d = nt * 16 + ln;
            size_t idx = ((size_t)h << 20) + ((size_t)q << 10) +
                         ((size_t)t << 6) + (size_t)d;
            AO[idx] = (_Float16)(o[nt][r] * inv);
        }
    }
}

// ---------------------------------------------------------------- launch
extern "C" void kernel_launch(void* const* d_in, const int* in_sizes, int n_in,
                              void* d_out, int out_size, void* d_ws,
                              size_t ws_size, hipStream_t stream) {
    (void)in_sizes; (void)n_in; (void)out_size; (void)ws_size;

    const float* x    = (const float*)d_in[0];  // 1*16*1024*512
    const float* Wqkv = (const float*)d_in[1];  // 512 x 1536
    const float* bqkv = (const float*)d_in[2];  // 1536
    const float* Wout = (const float*)d_in[3];  // 512 x 512
    const float* bout = (const float*)d_in[4];  // 512
    float* out = (float*)d_out;                 // 16384 x 512 f32

    const int M  = 16 * 1024;   // 16384 rows
    const int K  = 512;
    const int N1 = 1536;

    char* ws = (char*)d_ws;
    size_t off = 0;
    _Float16* Xh    = (_Float16*)(ws + off); off += (size_t)M * K * 2;    // 16 MB
    _Float16* WqkvT = (_Float16*)(ws + off); off += (size_t)N1 * K * 2;   // 1.5 MB
    _Float16* WoutT = (_Float16*)(ws + off); off += (size_t)K * K * 2;    // 0.5 MB
    _Float16* QKVh  = (_Float16*)(ws + off); off += (size_t)M * N1 * 2;   // 48 MB
    _Float16* AO    = (_Float16*)(ws + off); off += (size_t)M * K * 2;    // 16 MB

    // x -> f16
    {
        int n4 = (M * K) / 4;
        cvt_f16x4_kernel<<<(n4 + 255) / 256, 256, 0, stream>>>(
            (const float4*)x, (h4*)Xh, n4);
    }
    // weights -> transposed f16
    cvt_transpose_kernel<<<(K * N1 + 255) / 256, 256, 0, stream>>>(
        Wqkv, WqkvT, K, N1);
    cvt_transpose_kernel<<<(K * K + 255) / 256, 256, 0, stream>>>(
        Wout, WoutT, K, K);

    // GEMM1: QKV = Xh @ Wqkv + bqkv   (f16 out)
    wmma_gemm_kernel<true><<<dim3(M / 128, N1 / 64), 128, 0, stream>>>(
        Xh, WqkvT, bqkv, (void*)QKVh, M, N1, K);

    // attention over HW per (h, t), flash-style; writes AO in "bug" order
    attn_kernel<<<dim3(HWDIM / 64, 16, 8), 128, 0, stream>>>(QKVh, AO);

    // GEMM2: out = AO @ Wout + bout   (f32 out)
    wmma_gemm_kernel<false><<<dim3(M / 128, K / 64), 128, 0, stream>>>(
        AO, WoutT, bout, (void*)out, M, K, K);
}
